// AMGSRN_6141803233747
// MI455X (gfx1250) — matrix operand order repaired and checked by reading, not compiled, not run on hardware
//
#include <hip/hip_runtime.h>

typedef __attribute__((ext_vector_type(16))) _Float16 v16h;
typedef __attribute__((ext_vector_type(8)))  _Float16 v8h;
typedef __attribute__((ext_vector_type(8)))  float    v8f;

#define BATCH 524288
#define N_GRIDS 32
#define GRID_D 64
#define HID 128
#define WPB 2            // waves per block (64 threads)
#define MT 4             // 16-row M-tiles per wave -> 64 points/wave
#define TILE_IN0 (16 * 64)    // layer-0 input tile (halves), row stride 64
#define TILE     (16 * HID)   // hidden tile (halves), row stride 128

// d_ws layout (halves): W0h [128*64] @0, W1h [128*128] @8192, W2h @24576, W3h @40960
#define W0_OFF 0
#define W1_OFF 8192
#define W2_OFF 24576
#define W3_OFF 40960
#define W_TOTAL 57344

// dynamic LDS: s_tm(512 f32) | s_w4(128 f32) | s_bias(512 f32) |
//              per wave: bufA(MT*TILE f16), bufB(MT*TILE f16)
#define SMEM_BYTES ((512 + 128 + 512) * 4 + WPB * 2 * MT * TILE * 2)

__global__ __launch_bounds__(256) void convert_weights(
    const float* __restrict__ W0, const float* __restrict__ W1,
    const float* __restrict__ W2, const float* __restrict__ W3,
    _Float16* __restrict__ out) {
  int i = blockIdx.x * blockDim.x + threadIdx.x;
  if (i >= W_TOTAL) return;
  float v;
  if (i < W1_OFF)      v = W0[i - W0_OFF];
  else if (i < W2_OFF) v = W1[i - W1_OFF];
  else if (i < W3_OFF) v = W2[i - W2_OFF];
  else                 v = W3[i - W3_OFF];
  out[i] = (_Float16)v;
}

// ---- WMMA fragment loaders -------------------------------------------------
// A (16x32 f16): lane l holds row m=l&15; elems 0-15 = K[kh*16..+15]
// (kh=l>>4), within 32-wide K block kb.
__device__ __forceinline__ v16h load_a(const _Float16* h, int stride, int m, int kh, int kb) {
  const _Float16* base = h + m * stride + kb * 32 + kh * 16;
  v8h lo = *(const v8h*)(base);
  v8h hi = *(const v8h*)(base + 8);
  return __builtin_shufflevector(lo, hi, 0,1,2,3,4,5,6,7,8,9,10,11,12,13,14,15);
}
// B (32x16 f16): lane l holds col n=l&15; elems 0-15 = K[kb*32 + kh*16 .. +15],
// contiguous in row-major W[n][k] (weights stored [fan_out][fan_in]).
__device__ __forceinline__ v16h load_b(const _Float16* W, int fan_in, int n, int kh, int kb) {
  return *(const v16h*)(W + (size_t)n * fan_in + kb * 32 + kh * 16);
}

__device__ __forceinline__ void lds_fence() {
  asm volatile("s_wait_dscnt 0" ::: "memory");
  __builtin_amdgcn_wave_barrier();
}

__device__ __forceinline__ void relu_store(_Float16* hout, const v8f& c, float bias,
                                           int n, int kh) {
  #pragma unroll
  for (int r = 0; r < 8; ++r) {
    float v = c[r] + bias;                 // row M = r + 8*kh
    v = v > 0.f ? v : 0.f;
    hout[(r + 8 * kh) * HID + n] = (_Float16)v;
  }
}

// One MLP layer on MT 16-point tiles: hout = relu(hin @ W^T + b).
// N is processed in two halves of 4 N-tiles; per K-block, the MT A fragments
// are loaded from LDS ONCE and reused across 4 B fragments (16 WMMAs).
template <int NKB>
__device__ __forceinline__ void mlp_layer_mt(const _Float16* hin, int in_stride, int in_tile,
                                             const _Float16* Wl, const float* sbias,
                                             _Float16* hout, int lane) {
  const int m  = lane & 15;
  const int kh = lane >> 4;
  for (int nh = 0; nh < 2; ++nh) {
    v8f c[MT][4];
    #pragma unroll
    for (int t = 0; t < MT; ++t)
      #pragma unroll
      for (int j = 0; j < 4; ++j) c[t][j] = (v8f){};
    #pragma unroll
    for (int kb = 0; kb < NKB; ++kb) {
      v16h a[MT];
      #pragma unroll
      for (int t = 0; t < MT; ++t)
        a[t] = load_a(hin + t * in_tile, in_stride, m, kh, kb);
      #pragma unroll
      for (int j = 0; j < 4; ++j) {
        const int n = nh * 64 + j * 16 + m;   // C/D layout: N = lane&15
        v16h b = load_b(Wl, NKB * 32, n, kh, kb);
        #pragma unroll
        for (int t = 0; t < MT; ++t)
          c[t][j] = __builtin_amdgcn_wmma_f32_16x16x32_f16(
              false, a[t], false, b, (short)0, c[t][j], false, false);
      }
    }
    #pragma unroll
    for (int j = 0; j < 4; ++j) {
      const int n = nh * 64 + j * 16 + m;
      const float bias = sbias[n];
      #pragma unroll
      for (int t = 0; t < MT; ++t)
        relu_store(hout + t * TILE, c[t][j], bias, n, kh);
    }
  }
  lds_fence();
}

__global__ __launch_bounds__(WPB * 32) void amgsrn_fused(
    const float* __restrict__ x, const float* __restrict__ tm,
    const float* __restrict__ grids, const _Float16* __restrict__ Wh,
    const float* __restrict__ W4,
    const float* __restrict__ b0, const float* __restrict__ b1,
    const float* __restrict__ b2, const float* __restrict__ b3,
    const float* __restrict__ b4, float* __restrict__ y) {
  extern __shared__ char smem_raw[];
  float*    s_tm   = (float*)smem_raw;                     // 512 f32
  float*    s_w4   = s_tm + 512;                           // 128 f32
  float*    s_bias = s_w4 + 128;                           // 4*128 f32
  _Float16* bufs   = (_Float16*)(s_bias + 512);

  const int t = threadIdx.x;
  for (int i = t; i < N_GRIDS * 16; i += WPB * 32) s_tm[i] = tm[i];
  for (int i = t; i < HID; i += WPB * 32) {
    s_w4[i] = W4[i];
    s_bias[0 * HID + i] = b0[i];
    s_bias[1 * HID + i] = b1[i];
    s_bias[2 * HID + i] = b2[i];
    s_bias[3 * HID + i] = b3[i];
  }
  __syncthreads();

  const int wave = t >> 5;
  const int lane = t & 31;
  const int m    = lane & 15;
  const int kh   = lane >> 4;
  _Float16* hA = bufs + wave * 2 * MT * TILE;
  _Float16* hB = hA + MT * TILE;

  // this wave owns 64 consecutive points
  const int pbase = (blockIdx.x * WPB + wave) * (MT * 16);

  // ---- featurize MT tiles: lane handles (point m, grids of parity kh) ----
  for (int tt = 0; tt < MT; ++tt) {
    const int p = pbase + tt * 16 + m;
    const float px = x[p * 3 + 0], py = x[p * 3 + 1], pz = x[p * 3 + 2];
    _Float16* ft = hA + tt * TILE_IN0;
    for (int gp = 0; gp < N_GRIDS / 2; ++gp) {
      const int g = gp * 2 + kh;
      const float* M = &s_tm[g * 16];
      const float tpx = M[0]*px + M[1]*py + M[2] *pz + M[3];
      const float tpy = M[4]*px + M[5]*py + M[6] *pz + M[7];
      const float tpz = M[8]*px + M[9]*py + M[10]*pz + M[11];
      const float fx = (tpx + 1.f) * 0.5f * (GRID_D - 1);
      const float fy = (tpy + 1.f) * 0.5f * (GRID_D - 1);
      const float fz = (tpz + 1.f) * 0.5f * (GRID_D - 1);
      const float x0f = floorf(fx), y0f = floorf(fy), z0f = floorf(fz);
      const float tx = fx - x0f, ty = fy - y0f, tz = fz - z0f;
      const int ix0 = (int)x0f, iy0 = (int)y0f, iz0 = (int)z0f;
      const float* gb = grids + (size_t)g * 2 * (GRID_D * GRID_D * GRID_D);
      float f0 = 0.f, f1 = 0.f;
      #pragma unroll
      for (int cor = 0; cor < 8; ++cor) {
        const int dx = cor & 1, dy = (cor >> 1) & 1, dz = cor >> 2;
        const int xi = ix0 + dx, yi = iy0 + dy, zi = iz0 + dz;
        float w = (dx ? tx : 1.f - tx) * (dy ? ty : 1.f - ty) * (dz ? tz : 1.f - tz);
        const bool valid = (xi >= 0) & (xi < GRID_D) & (yi >= 0) & (yi < GRID_D)
                         & (zi >= 0) & (zi < GRID_D);
        w = valid ? w : 0.f;
        int xc = xi < 0 ? 0 : (xi > GRID_D - 1 ? GRID_D - 1 : xi);
        int yc = yi < 0 ? 0 : (yi > GRID_D - 1 ? GRID_D - 1 : yi);
        int zc = zi < 0 ? 0 : (zi > GRID_D - 1 ? GRID_D - 1 : zi);
        const int off = (zc * GRID_D + yc) * GRID_D + xc;
        f0 += gb[off] * w;
        f1 += gb[GRID_D * GRID_D * GRID_D + off] * w;
      }
      ft[m * 64 + 2 * g + 0] = (_Float16)f0;   // feature k = g*2 + c
      ft[m * 64 + 2 * g + 1] = (_Float16)f1;
    }
  }
  lds_fence();

  // ---- MLP: 64 -> 128 -> 128 -> 128 -> 128 (ReLU), LDS ping-pong ----
  mlp_layer_mt<2>(hA, 64,  TILE_IN0, Wh + W0_OFF, s_bias + 0 * HID, hB, lane);
  mlp_layer_mt<4>(hB, HID, TILE,     Wh + W1_OFF, s_bias + 1 * HID, hA, lane);
  mlp_layer_mt<4>(hA, HID, TILE,     Wh + W2_OFF, s_bias + 2 * HID, hB, lane);
  mlp_layer_mt<4>(hB, HID, TILE,     Wh + W3_OFF, s_bias + 3 * HID, hA, lane);

  // ---- final layer 128 -> 1 (VALU): split K between lane halves ----
  const float bias4 = b4[0];
  for (int tt = 0; tt < MT; ++tt) {
    const _Float16* ht = hA + tt * TILE;
    const int k0 = kh * 64;
    const v8h* hv = (const v8h*)(ht + m * HID + k0);   // 64 halves = 8 x v8h
    float acc = 0.f;
    #pragma unroll
    for (int c8 = 0; c8 < 8; ++c8) {
      v8h hc = hv[c8];
      #pragma unroll
      for (int e = 0; e < 8; ++e)
        acc += (float)hc[e] * s_w4[k0 + c8 * 8 + e];
    }
    acc += __shfl_xor(acc, 16, 32);
    if (kh == 0) y[pbase + tt * 16 + m] = acc + bias4;   // VOL scale = *1 + 0
  }
}

extern "C" void kernel_launch(void* const* d_in, const int* in_sizes, int n_in,
                              void* d_out, int out_size, void* d_ws, size_t ws_size,
                              hipStream_t stream) {
  const float* x     = (const float*)d_in[0];
  const float* tm    = (const float*)d_in[1];
  const float* grids = (const float*)d_in[2];
  const float* W0 = (const float*)d_in[3];  const float* b0 = (const float*)d_in[4];
  const float* W1 = (const float*)d_in[5];  const float* b1 = (const float*)d_in[6];
  const float* W2 = (const float*)d_in[7];  const float* b2 = (const float*)d_in[8];
  const float* W3 = (const float*)d_in[9];  const float* b3 = (const float*)d_in[10];
  const float* W4 = (const float*)d_in[11]; const float* b4 = (const float*)d_in[12];
  float* y = (float*)d_out;
  _Float16* Wh = (_Float16*)d_ws;

  convert_weights<<<(W_TOTAL + 255) / 256, 256, 0, stream>>>(W0, W1, W2, W3, Wh);
  // 524288 points / (64 per wave * 2 waves per block) = 4096 blocks
  amgsrn_fused<<<BATCH / (MT * 16 * WPB), WPB * 32, SMEM_BYTES, stream>>>(
      x, tm, grids, Wh, W4, b0, b1, b2, b3, b4, y);
}
